// DinAttentionLayer_63136019251872
// MI455X (gfx1250) — matrix operand order repaired and checked by reading, not compiled, not run on hardware
//
#include <hip/hip_runtime.h>

typedef __attribute__((ext_vector_type(16))) _Float16 v16h;
typedef __attribute__((ext_vector_type(8)))  _Float16 v8h;
typedef __attribute__((ext_vector_type(8)))  float    v8f;

union V16H { v16h v; v8h h[2]; };

// B=64, T=1024, M=256, D=1024, K(=4M)=1024

__device__ inline v8f wmma16(v16h a, v16h b, v8f c) {
  // v_wmma_f32_16x16x32_f16 : D = A(16x32 f16) x B(32x16 f16) + C(16x16 f32)
  return __builtin_amdgcn_wmma_f32_16x16x32_f16(
      /*neg_a=*/false, a, /*neg_b=*/false, b,
      /*c_mod=*/(short)0, c, /*reuse_a=*/false, /*reuse_b=*/false);
}

// ---------------------------------------------------------------------------
// Prep: out[n*1024 + k] = (f16) in[k*1024 + n]   (1024x1024)
// Puts weights in [N][K] order so a WMMA B-fragment (lane n, 16 consecutive K)
// is one contiguous 32-byte load.
// ---------------------------------------------------------------------------
__global__ __launch_bounds__(256) void wtrans(const float* __restrict__ in,
                                              _Float16* __restrict__ out) {
  __shared__ float tile[32][33];
  const int kb = blockIdx.x * 32;
  const int nb = blockIdx.y * 32;
  const int tx = threadIdx.x & 31;
  const int ty = threadIdx.x >> 5;  // 0..7
#pragma unroll
  for (int j = 0; j < 32; j += 8)
    tile[ty + j][tx] = in[(size_t)(kb + ty + j) * 1024 + (nb + tx)];
  __syncthreads();
#pragma unroll
  for (int j = 0; j < 32; j += 8)
    out[(size_t)(nb + ty + j) * 1024 + (kb + tx)] = (_Float16)tile[tx][ty + j];
}

// ---------------------------------------------------------------------------
// Main fused kernel: att_input build -> L1 GEMM+ReLU -> L2 GEMM+ReLU -> score
// One WG = 64 rows (one b, t0..t0+63). 512 threads = 16 waves.
// Each wave: 4 M-tiles x 4 N-tiles of 16x16 f32 accum (K-loop step 32).
// ---------------------------------------------------------------------------
__global__ __launch_bounds__(512) void din_gemm(
    const float* __restrict__ query, const float* __restrict__ keys,
    const _Float16* __restrict__ w1t, const _Float16* __restrict__ w2t,
    const float* __restrict__ b1, const float* __restrict__ b2,
    const float* __restrict__ wscore, float* __restrict__ scores) {
  extern __shared__ char smem[];
  _Float16* a_lds = (_Float16*)smem;                       // 64*1024 f16 = 128KB
  _Float16* h_lds = (_Float16*)(smem + 64 * 1024 * 2);     // 64*1024 f16 = 128KB
  float*    spart = (float*)(smem + 2 * 64 * 1024 * 2);    // 16 waves * 64 rows

  const int b   = blockIdx.x >> 4;
  const int t0  = (blockIdx.x & 15) * 64;
  const int tid = threadIdx.x;

  // ---- build att_input tile [64 rows x 1024] in f16: [q | k | q-k | q*k] ----
  for (int idx = tid; idx < 64 * 256; idx += 512) {
    const int row = idx >> 8;       // 0..63
    const int c   = idx & 255;      // 0..255
    const float qv = query[b * 256 + c];
    const float kv = keys[((size_t)(b * 1024 + t0 + row)) * 256 + c];
    _Float16* ar = a_lds + row * 1024;
    ar[c]       = (_Float16)qv;
    ar[256 + c] = (_Float16)kv;
    ar[512 + c] = (_Float16)(qv - kv);
    ar[768 + c] = (_Float16)(qv * kv);
  }
  __syncthreads();

  const int lane = tid & 31;   // wave32
  const int wave = tid >> 5;   // 0..15
  const int l15  = lane & 15;
  const int hsel = lane >> 4;  // 0: K-low half lanes, 1: K-high half lanes

  // ======================= layer 1: h = relu(A @ W1 + b1) ====================
  {
    v8f acc[4][4] = {};
    for (int k0 = 0; k0 < 1024; k0 += 32) {
      v16h afrag[4];
      const int kA = k0 + hsel * 16;  // A layout: lane m, 16 consecutive K; lanes16-31 +16
#pragma unroll
      for (int mt = 0; mt < 4; ++mt) {
        const _Float16* ap = a_lds + (mt * 16 + l15) * 1024 + kA;
        V16H u;
        u.h[0] = *(const v8h*)(ap);
        u.h[1] = *(const v8h*)(ap + 8);
        afrag[mt] = u.v;
      }
      const int kB = k0 + hsel * 16;  // B layout: lane n, 16 consecutive K
#pragma unroll
      for (int nt = 0; nt < 4; ++nt) {
        const int n = (wave * 4 + nt) * 16 + l15;
        v16h bfrag = *(const v16h*)(w1t + (size_t)n * 1024 + kB);
#pragma unroll
        for (int mt = 0; mt < 4; ++mt)
          acc[mt][nt] = wmma16(afrag[mt], bfrag, acc[mt][nt]);
      }
    }
    // bias + relu + store h tile (row-major f16, A-fragment friendly)
#pragma unroll
    for (int nt = 0; nt < 4; ++nt) {
      const int n = (wave * 4 + nt) * 16 + l15;
      const float bias = b1[n];
#pragma unroll
      for (int mt = 0; mt < 4; ++mt) {
#pragma unroll
        for (int r = 0; r < 8; ++r) {
          float v = acc[mt][nt][r] + bias;
          v = v > 0.f ? v : 0.f;
          const int m = mt * 16 + r + hsel * 8;  // C/D layout: lanes16-31 hold M+8
          h_lds[m * 1024 + n] = (_Float16)v;
        }
      }
    }
  }
  __syncthreads();

  // ============ layer 2: score = relu(h @ W2 + b2) . w_score ================
  {
    v8f acc[4][4] = {};
    for (int k0 = 0; k0 < 1024; k0 += 32) {
      v16h afrag[4];
      const int kA = k0 + hsel * 16;
#pragma unroll
      for (int mt = 0; mt < 4; ++mt) {
        const _Float16* ap = h_lds + (mt * 16 + l15) * 1024 + kA;
        V16H u;
        u.h[0] = *(const v8h*)(ap);
        u.h[1] = *(const v8h*)(ap + 8);
        afrag[mt] = u.v;
      }
      const int kB = k0 + hsel * 16;
#pragma unroll
      for (int nt = 0; nt < 4; ++nt) {
        const int n = (wave * 4 + nt) * 16 + l15;
        v16h bfrag = *(const v16h*)(w2t + (size_t)n * 1024 + kB);
#pragma unroll
        for (int mt = 0; mt < 4; ++mt)
          acc[mt][nt] = wmma16(afrag[mt], bfrag, acc[mt][nt]);
      }
    }
    // per-lane partial of relu(h2)*w_score over this wave's 64 n-columns
    float sacc[4][8];
#pragma unroll
    for (int mt = 0; mt < 4; ++mt)
#pragma unroll
      for (int r = 0; r < 8; ++r) sacc[mt][r] = 0.f;
#pragma unroll
    for (int nt = 0; nt < 4; ++nt) {
      const int n = (wave * 4 + nt) * 16 + l15;
      const float bias = b2[n];
      const float ws   = wscore[n];
#pragma unroll
      for (int mt = 0; mt < 4; ++mt) {
#pragma unroll
        for (int r = 0; r < 8; ++r) {
          float v = acc[mt][nt][r] + bias;
          v = v > 0.f ? v : 0.f;
          sacc[mt][r] += v * ws;
        }
      }
    }
    // reduce the 16 n-lanes of each half-wave; deterministic per-wave partials
#pragma unroll
    for (int mt = 0; mt < 4; ++mt) {
#pragma unroll
      for (int r = 0; r < 8; ++r) {
        float v = sacc[mt][r];
        v += __shfl_xor(v, 1, 16);
        v += __shfl_xor(v, 2, 16);
        v += __shfl_xor(v, 4, 16);
        v += __shfl_xor(v, 8, 16);
        if (l15 == 0) {
          const int m = mt * 16 + r + hsel * 8;  // lane0 -> m, lane16 -> m+8
          spart[wave * 64 + m] = v;
        }
      }
    }
  }
  __syncthreads();

  if (tid < 64) {
    float s = 0.f;
#pragma unroll
    for (int w = 0; w < 16; ++w) s += spart[w * 64 + tid];
    scores[(size_t)b * 1024 + t0 + tid] = s;
  }
}

// ---------------------------------------------------------------------------
// Masked softmax over T, then out[b, dseg] = sum_t p[t] * values[b,t,dseg]
// grid (B, D/256), block 256. Each values element read exactly once.
// ---------------------------------------------------------------------------
__global__ __launch_bounds__(256) void softmax_av(
    const float* __restrict__ scores, const float* __restrict__ mask,
    const float* __restrict__ values, const float* __restrict__ b_score,
    float* __restrict__ out) {
  __shared__ float p[1024];
  __shared__ float red[256];
  const int b     = blockIdx.x;
  const int dbase = blockIdx.y * 256;
  const int tid   = threadIdx.x;
  const float bs  = b_score[0];

  float lmax = -3.4e38f;
#pragma unroll
  for (int i = 0; i < 4; ++i) {
    const int t = tid + i * 256;
    const float l = scores[(size_t)b * 1024 + t] + bs +
                    mask[(size_t)b * 1024 + t] * (-1e9f);
    p[t] = l;
    lmax = fmaxf(lmax, l);
  }
  red[tid] = lmax;
  __syncthreads();
  for (int s = 128; s > 0; s >>= 1) {
    if (tid < s) red[tid] = fmaxf(red[tid], red[tid + s]);
    __syncthreads();
  }
  const float mx = red[0];
  __syncthreads();

  float lsum = 0.f;
#pragma unroll
  for (int i = 0; i < 4; ++i) {
    const int t = tid + i * 256;
    const float e = __expf(p[t] - mx);
    p[t] = e;
    lsum += e;
  }
  red[tid] = lsum;
  __syncthreads();
  for (int s = 128; s > 0; s >>= 1) {
    if (tid < s) red[tid] += red[tid + s];
    __syncthreads();
  }
  const float inv = 1.f / red[0];

  const int d = dbase + tid;
  const float* vcol = values + (size_t)b * 1024 * 1024 + d;
  float acc = 0.f;
#pragma unroll 4
  for (int t = 0; t < 1024; ++t)
    acc = fmaf(p[t], vcol[(size_t)t * 1024], acc);
  out[(size_t)b * 1024 + d] = acc * inv;
}

// ---------------------------------------------------------------------------
extern "C" void kernel_launch(void* const* d_in, const int* in_sizes, int n_in,
                              void* d_out, int out_size, void* d_ws, size_t ws_size,
                              hipStream_t stream) {
  (void)in_sizes; (void)n_in; (void)out_size; (void)ws_size;
  const float* query  = (const float*)d_in[0];
  const float* keys   = (const float*)d_in[1];
  const float* values = (const float*)d_in[2];
  const float* mask   = (const float*)d_in[3];
  const float* W1     = (const float*)d_in[4];
  const float* b1     = (const float*)d_in[5];
  const float* W2     = (const float*)d_in[6];
  const float* b2     = (const float*)d_in[7];
  const float* wsc    = (const float*)d_in[8];
  const float* bsc    = (const float*)d_in[9];

  char* ws = (char*)d_ws;
  _Float16* w1t  = (_Float16*)ws;                                // 2 MB
  _Float16* w2t  = (_Float16*)(ws + (size_t)1024 * 1024 * 2);    // 2 MB
  float*    scrs = (float*)(ws + (size_t)2 * 1024 * 1024 * 2);   // 256 KB
  float*    out  = (float*)d_out;

  wtrans<<<dim3(32, 32), 256, 0, stream>>>(W1, w1t);
  wtrans<<<dim3(32, 32), 256, 0, stream>>>(W2, w2t);

  const size_t lds_bytes = (size_t)2 * 64 * 1024 * 2 + 16 * 64 * 4;  // 260 KB
  (void)hipFuncSetAttribute((const void*)din_gemm,
                            hipFuncAttributeMaxDynamicSharedMemorySize,
                            (int)lds_bytes);
  din_gemm<<<1024, 512, lds_bytes, stream>>>(query, keys, w1t, w2t, b1, b2,
                                             wsc, scrs);

  softmax_av<<<dim3(64, 4), 256, 0, stream>>>(scrs, mask, values, bsc, out);
}
